// OneDParallelMHA_77850577207464
// MI455X (gfx1250) — compile-verified, matrix operand order checked by reading, not run on hardware
//
#include <hip/hip_runtime.h>
#include <hip/hip_bf16.h>

// ---------------------------------------------------------------------------
// MI455X (gfx1250) fused MHA: bf16 convert -> QKV GEMM -> RoPE -> flash attn
// -> out GEMM. All matmuls on v_wmma_f32_16x16x32_bf16 (fp32 accumulate).
// LDS staging uses GLOBAL_LOAD_ASYNC_TO_LDS_B128 when the toolchain exposes
// the gfx1250 builtins (guarded by __has_builtin, with a b128 copy fallback).
// ---------------------------------------------------------------------------

typedef __bf16 bf16_t;
typedef bf16_t v8bf  __attribute__((ext_vector_type(8)));
typedef bf16_t v16bf __attribute__((ext_vector_type(16)));
typedef float  v4f   __attribute__((ext_vector_type(4)));
typedef float  v8f   __attribute__((ext_vector_type(8)));
typedef int    v4i   __attribute__((ext_vector_type(4)));

#define B_DIM 2
#define S_LEN 2048
#define E_DIM 2048
#define HEADS 16
#define DHEAD 128

#if defined(__HIP_DEVICE_COMPILE__) &&                                        \
    __has_builtin(__builtin_amdgcn_global_load_async_to_lds_b128) &&          \
    __has_builtin(__builtin_amdgcn_s_wait_asynccnt)
#define USE_ASYNC_COPY 1
#else
#define USE_ASYNC_COPY 0
#endif

#define AS1 __attribute__((address_space(1)))
#define AS3 __attribute__((address_space(3)))

// Copy 16 bytes global -> LDS. Async (no VGPR round-trip, ASYNCcnt) if the
// gfx1250 builtin exists, otherwise b128 load + ds_store_b128.
// Builtin signature (from clang diagnostic): (AS1 int4*, AS3 int4*, imm, imm).
__device__ __forceinline__ void copy16_g2l(const bf16_t* g, bf16_t* l) {
#if USE_ASYNC_COPY
  __builtin_amdgcn_global_load_async_to_lds_b128((AS1 v4i*)g, (AS3 v4i*)l,
                                                 0, 0);
#else
  *(v8bf*)l = *(const v8bf*)g;
#endif
}

__device__ __forceinline__ void wait_async_then_barrier() {
#if USE_ASYNC_COPY
  __builtin_amdgcn_s_wait_asynccnt(0);
#endif
  __syncthreads();
}

// 16x32 bf16 fragment from LDS in the CDNA5 WMMA 16-bit A/B lane layout:
// lane: row = lane%16, khalf = lane/16
// elems 0..7  -> k = khalf*8 + j       (contiguous 16B)
// elems 8..15 -> k = 16 + khalf*8 + j  (contiguous 16B)
__device__ __forceinline__ v16bf load_frag_b(const bf16_t* base, int ld) {
  const int lane  = threadIdx.x & 31;
  const int r     = lane & 15;
  const int half8 = (lane >> 4) << 3;
  const bf16_t* p = base + r * ld + half8;
  v8bf lo = *(const v8bf*)(p);
  v8bf hi = *(const v8bf*)(p + 16);
  return __builtin_shufflevector(lo, hi, 0, 1, 2, 3, 4, 5, 6, 7,
                                 8, 9, 10, 11, 12, 13, 14, 15);
}

__device__ __forceinline__ v8f wmma_bf16(v16bf a, v16bf b, v8f c) {
  return __builtin_amdgcn_wmma_f32_16x16x32_bf16(
      false, a, false, b, (short)0, c, false, false);
}

// ---------------------------------------------------------------------------
// fp32 -> bf16 bulk conversion (one-time; makes every GEMM operand bf16).
// ---------------------------------------------------------------------------
__global__ __launch_bounds__(256) void conv_f32_to_bf16(
    const float* __restrict__ src, bf16_t* __restrict__ dst, size_t n) {
  const size_t i = ((size_t)blockIdx.x * 256 + threadIdx.x) * 8;
  if (i >= n) return;
  v4f a = *(const v4f*)(src + i);
  v4f b = *(const v4f*)(src + i + 4);
  v8bf o;
  o[0] = (bf16_t)a[0]; o[1] = (bf16_t)a[1]; o[2] = (bf16_t)a[2];
  o[3] = (bf16_t)a[3]; o[4] = (bf16_t)b[0]; o[5] = (bf16_t)b[1];
  o[6] = (bf16_t)b[2]; o[7] = (bf16_t)b[3];
  *(v8bf*)(dst + i) = o;
}

__device__ __forceinline__ void store_out(float* C, size_t i, float v) {
  C[i] = v;
}
__device__ __forceinline__ void store_out(bf16_t* C, size_t i, float v) {
  C[i] = (bf16_t)v;
}

// ---------------------------------------------------------------------------
// C[M,N] = A[M,K] @ W[N,K]^T + bias[N], A/W bf16 row-major.
// 128x128 block, BK=32, 256 threads = 8 waves; wave tile 64x32 (8 WMMA/step).
// Double-buffered LDS; next tile async-prefetched while current computes.
// ---------------------------------------------------------------------------
template <typename OutT>
__global__ __launch_bounds__(256) void gemm_bf16_wmma(
    const bf16_t* __restrict__ A, const bf16_t* __restrict__ W,
    const float* __restrict__ bias, OutT* __restrict__ C,
    int M, int N, int K) {
  __shared__ __align__(16) bf16_t As[2][128 * 32];
  __shared__ __align__(16) bf16_t Ws[2][128 * 32];

  const int row0  = blockIdx.y * 128;
  const int col0  = blockIdx.x * 128;
  const int wave  = threadIdx.x >> 5;   // 0..7
  const int waveM = (wave >> 2) * 64;   // 0 / 64
  const int waveN = (wave & 3) * 32;    // 0/32/64/96
  const int t     = threadIdx.x;

  v8f acc[4][2] = {};

  // stage one 128x32 bf16 tile pair; 512 16B-chunks per tile, 2 per thread.
  auto stage = [&](int buf, int k0) {
#pragma unroll
    for (int c = 0; c < 2; ++c) {
      const int chunk = t + c * 256;
      const int r  = chunk >> 2;
      const int ko = (chunk & 3) * 8;
      copy16_g2l(A + (size_t)(row0 + r) * K + k0 + ko, &As[buf][r * 32 + ko]);
      copy16_g2l(W + (size_t)(col0 + r) * K + k0 + ko, &Ws[buf][r * 32 + ko]);
    }
  };

  stage(0, 0);
  int buf = 0;
  for (int k0 = 0; k0 < K; k0 += 32) {
    wait_async_then_barrier();
    if (k0 + 32 < K) stage(buf ^ 1, k0 + 32);

    v16bf af[4], wf[2];
#pragma unroll
    for (int i = 0; i < 4; ++i)
      af[i] = load_frag_b(&As[buf][(waveM + i * 16) * 32], 32);
#pragma unroll
    for (int j = 0; j < 2; ++j)
      wf[j] = load_frag_b(&Ws[buf][(waveN + j * 16) * 32], 32);
#pragma unroll
    for (int i = 0; i < 4; ++i)
#pragma unroll
      for (int j = 0; j < 2; ++j) acc[i][j] = wmma_bf16(af[i], wf[j], acc[i][j]);
    buf ^= 1;
  }

  // C/D layout: lane holds col n = lane%16; VGPR r -> row r + 8*(lane/16)
  const int lane = threadIdx.x & 31;
  const int n    = lane & 15;
  const int rsel = (lane >> 4) << 3;
#pragma unroll
  for (int ti = 0; ti < 4; ++ti)
#pragma unroll
    for (int tj = 0; tj < 2; ++tj) {
      const int col  = col0 + waveN + tj * 16 + n;
      const float bv = bias[col];
#pragma unroll
      for (int r = 0; r < 8; ++r) {
        const int row = row0 + waveM + ti * 16 + rsel + r;
        store_out(C, (size_t)row * N + col, acc[ti][tj][r] + bv);
      }
    }
}

// ---------------------------------------------------------------------------
// RoPE in-place on bf16 qkv[B,S,3,H,D] (fp32 math, bf16 storage).
// ---------------------------------------------------------------------------
__global__ __launch_bounds__(256) void rope_kernel(bf16_t* __restrict__ qkv) {
  const size_t idx   = (size_t)blockIdx.x * blockDim.x + threadIdx.x;
  const size_t total = (size_t)B_DIM * S_LEN * HEADS * 64;
  if (idx >= total) return;
  const int i = (int)(idx & 63);
  const int h = (int)((idx >> 6) & 15);
  const int s = (int)((idx >> 10) & 2047);
  const int b = (int)(idx >> 21);
  const float inv_freq = __expf(-(float)i * 0.14391156831212787f);  // ln(1e4)/64
  const float fr = (float)s * inv_freq;
  float sn, c;
  __sincosf(fr, &sn, &c);
#pragma unroll
  for (int split = 0; split < 2; ++split) {
    const size_t base =
        ((((size_t)b * S_LEN + s) * 3 + split) * HEADS + h) * DHEAD;
    const float x1 = (float)qkv[base + i];
    const float x2 = (float)qkv[base + 64 + i];
    qkv[base + i]      = (bf16_t)(x1 * c - x2 * sn);
    qkv[base + 64 + i] = (bf16_t)(x1 * sn + x2 * c);
  }
}

// ---------------------------------------------------------------------------
// Causal flash attention: grid (S/64, H, B), 128 threads = 4 waves.
// bf16 q/k/v; K tile async-staged to LDS; V transposed manually; P via LDS.
// ---------------------------------------------------------------------------
__global__ __launch_bounds__(128) void flash_attn_wmma(
    const bf16_t* __restrict__ qkv, bf16_t* __restrict__ ctx) {
  __shared__ __align__(16) bf16_t Kt[64 * 128];    // (j, d) row-major
  __shared__ __align__(16) bf16_t Vt[128 * 64];    // (d, j) transposed
  __shared__ __align__(16) bf16_t Ps[4 * 16 * 64]; // per-wave (m, j)

  const int qt   = blockIdx.x;
  const int h    = blockIdx.y;
  const int b    = blockIdx.z;
  const int wave = threadIdx.x >> 5;
  const int lane = threadIdx.x & 31;
  const int n    = lane & 15;
  const int rsel = (lane >> 4) << 3;
  const float scale = 0.08838834764831845f;  // 1/sqrt(128)
  const size_t ROW = (size_t)3 * HEADS * DHEAD;  // qkv row stride (per s)

  // wave's 16 Q rows as 4 fragments (unscaled; scale applied in softmax)
  v16bf qf[4];
  {
    const int sq = qt * 64 + wave * 16 + n;
    const bf16_t* qrow = qkv + ((size_t)b * S_LEN + sq) * ROW + (size_t)h * DHEAD;
#pragma unroll
    for (int kk = 0; kk < 4; ++kk) {
      const bf16_t* p = qrow + kk * 32 + rsel;
      v8bf lo = *(const v8bf*)(p);
      v8bf hi = *(const v8bf*)(p + 16);
      qf[kk] = __builtin_shufflevector(lo, hi, 0, 1, 2, 3, 4, 5, 6, 7,
                                       8, 9, 10, 11, 12, 13, 14, 15);
    }
  }

  v8f cacc[8] = {};
  float mrow[8], lrow[8];
#pragma unroll
  for (int r = 0; r < 8; ++r) { mrow[r] = -1e30f; lrow[r] = 0.f; }

  for (int jt = 0; jt <= qt; ++jt) {
    __syncthreads();  // previous iteration's LDS reads complete
    {
      const size_t kvbase = ((size_t)b * S_LEN + jt * 64) * ROW;
      const bf16_t* kbase = qkv + kvbase + (size_t)(HEADS + h) * DHEAD;
      // K tile: 64x128 bf16 = 1024 16B chunks, 8 per thread (async path)
#pragma unroll
      for (int c = 0; c < 8; ++c) {
        const int chunk = threadIdx.x + c * 128;
        const int j   = chunk >> 4;
        const int off = (chunk & 15) * 8;
        copy16_g2l(kbase + (size_t)j * ROW + off, &Kt[j * 128 + off]);
      }
      // V tile transposed: thread owns (row j, 64-col half), scatter b16
      const int j  = threadIdx.x >> 1;
      const int dh = (threadIdx.x & 1) * 64;
      const bf16_t* vrow =
          qkv + kvbase + (size_t)j * ROW + (size_t)(2 * HEADS + h) * DHEAD + dh;
#pragma unroll
      for (int c8 = 0; c8 < 8; ++c8) {
        v8bf vv = *(const v8bf*)(vrow + c8 * 8);
#pragma unroll
        for (int e = 0; e < 8; ++e) Vt[(dh + c8 * 8 + e) * 64 + j] = vv[e];
      }
    }
    wait_async_then_barrier();

    // S = Q @ K^T : wave computes 16x64 (4 col tiles x 4 k-steps)
    v8f sacc[4] = {};
#pragma unroll
    for (int nt = 0; nt < 4; ++nt) {
#pragma unroll
      for (int kk = 0; kk < 4; ++kk) {
        v16bf bfK = load_frag_b(Kt + (nt * 16) * 128 + kk * 32, 128);
        sacc[nt] = wmma_bf16(qf[kk], bfK, sacc[nt]);
      }
    }

    // scaled scores + causal mask (diagonal tile only)
    float sv[4][8];
#pragma unroll
    for (int nt = 0; nt < 4; ++nt)
#pragma unroll
      for (int r = 0; r < 8; ++r) sv[nt][r] = sacc[nt][r] * scale;
    if (jt == qt) {
#pragma unroll
      for (int nt = 0; nt < 4; ++nt) {
        const int col = nt * 16 + n;
#pragma unroll
        for (int r = 0; r < 8; ++r)
          if (col > wave * 16 + rsel + r) sv[nt][r] = -1e30f;
      }
    }

    // online softmax (fp32 stats; reductions inside 16-lane halves)
    float pj[4][8];
#pragma unroll
    for (int r = 0; r < 8; ++r) {
      float vmax = fmaxf(fmaxf(sv[0][r], sv[1][r]), fmaxf(sv[2][r], sv[3][r]));
#pragma unroll
      for (int m = 1; m < 16; m <<= 1)
        vmax = fmaxf(vmax, __shfl_xor(vmax, m, 32));
      const float mnew = fmaxf(mrow[r], vmax);
      const float cf = __expf(mrow[r] - mnew);
      float rs = 0.f;
#pragma unroll
      for (int nt = 0; nt < 4; ++nt) {
        const float p = __expf(sv[nt][r] - mnew);
        pj[nt][r] = p;
        rs += p;
      }
#pragma unroll
      for (int m = 1; m < 16; m <<= 1) rs += __shfl_xor(rs, m, 32);
      lrow[r] = lrow[r] * cf + rs;
      mrow[r] = mnew;
#pragma unroll
      for (int dt = 0; dt < 8; ++dt) cacc[dt][r] *= cf;
    }

    // stage P (bf16): C-layout -> A-layout via LDS
    bf16_t* pw = Ps + wave * 16 * 64;
#pragma unroll
    for (int nt = 0; nt < 4; ++nt)
#pragma unroll
      for (int r = 0; r < 8; ++r)
        pw[(rsel + r) * 64 + nt * 16 + n] = (bf16_t)pj[nt][r];
    __syncthreads();

    // ctx(16x128) += P(16x64) @ V(64x128)
#pragma unroll
    for (int kk = 0; kk < 2; ++kk) {
      v16bf af = load_frag_b(pw + kk * 32, 64);
#pragma unroll
      for (int dt = 0; dt < 8; ++dt) {
        v16bf bfV = load_frag_b(Vt + (dt * 16) * 64 + kk * 32, 64);
        cacc[dt] = wmma_bf16(af, bfV, cacc[dt]);
      }
    }
  }

  // normalize, write ctx[b, s, h*128+d] as bf16 (feeds the out-proj GEMM)
  {
    const int sq = qt * 64 + wave * 16 + rsel;
#pragma unroll
    for (int r = 0; r < 8; ++r) {
      const float inv = 1.f / lrow[r];
      const size_t base = ((size_t)b * S_LEN + (sq + r)) * E_DIM + h * DHEAD;
#pragma unroll
      for (int dt = 0; dt < 8; ++dt)
        ctx[base + dt * 16 + n] = (bf16_t)(cacc[dt][r] * inv);
    }
  }
}

// ---------------------------------------------------------------------------
extern "C" void kernel_launch(void* const* d_in, const int* in_sizes, int n_in,
                              void* d_out, int out_size, void* d_ws,
                              size_t ws_size, hipStream_t stream) {
  const float* x      = (const float*)d_in[0];
  const float* wqkv_w = (const float*)d_in[1];
  const float* wqkv_b = (const float*)d_in[2];
  const float* out_w  = (const float*)d_in[3];
  const float* out_b  = (const float*)d_in[4];
  float* out = (float*)d_out;

  const int M = B_DIM * S_LEN;  // 4096

  // workspace (bf16): x 8M, wqkv 12M, outw 4M, qkv 24M, ctx 8M elems = 112MB
  bf16_t* x_bf    = (bf16_t*)d_ws;
  bf16_t* wqkv_bf = x_bf + (size_t)M * E_DIM;
  bf16_t* outw_bf = wqkv_bf + (size_t)(3 * E_DIM) * E_DIM;
  bf16_t* qkv_bf  = outw_bf + (size_t)E_DIM * E_DIM;
  bf16_t* ctx_bf  = qkv_bf + (size_t)M * (3 * E_DIM);

  // 0) one-time fp32 -> bf16 conversions
  const size_t nx = (size_t)M * E_DIM;
  const size_t nw = (size_t)(3 * E_DIM) * E_DIM;
  const size_t no = (size_t)E_DIM * E_DIM;
  conv_f32_to_bf16<<<(unsigned)(nx / 8 / 256), 256, 0, stream>>>(x, x_bf, nx);
  conv_f32_to_bf16<<<(unsigned)(nw / 8 / 256), 256, 0, stream>>>(wqkv_w, wqkv_bf, nw);
  conv_f32_to_bf16<<<(unsigned)(no / 8 / 256), 256, 0, stream>>>(out_w, outw_bf, no);

  // 1) QKV = X @ Wqkv^T + b  (bf16 out)
  gemm_bf16_wmma<bf16_t><<<dim3((3 * E_DIM) / 128, M / 128), 256, 0, stream>>>(
      x_bf, wqkv_bf, wqkv_b, qkv_bf, M, 3 * E_DIM, E_DIM);

  // 2) RoPE on Q and K
  const size_t rope_threads = (size_t)B_DIM * S_LEN * HEADS * 64;
  rope_kernel<<<(unsigned)((rope_threads + 255) / 256), 256, 0, stream>>>(qkv_bf);

  // 3) causal flash attention -> ctx (bf16)
  flash_attn_wmma<<<dim3(S_LEN / 64, HEADS, B_DIM), 128, 0, stream>>>(qkv_bf,
                                                                      ctx_bf);

  // 4) out = ctx @ Wout^T + b  (fp32 out)
  gemm_bf16_wmma<float><<<dim3(E_DIM / 128, M / 128), 256, 0, stream>>>(
      ctx_bf, outw_bf, out_b, out, M, E_DIM, E_DIM);
}
